// WindowAttention_22479858827271
// MI455X (gfx1250) — matrix operand order repaired and checked by reading, not compiled, not run on hardware
//
#include <hip/hip_runtime.h>

typedef __attribute__((ext_vector_type(16))) __bf16 v16bf;
typedef __attribute__((ext_vector_type(8)))  float  v8f;
typedef unsigned short u16;
typedef unsigned int   u32;
typedef __attribute__((ext_vector_type(4))) unsigned int v4u;
typedef __attribute__((ext_vector_type(4))) int v4i;
typedef __attribute__((ext_vector_type(8))) int v8i;

#if __has_builtin(__builtin_amdgcn_tensor_load_to_lds) && __has_builtin(__builtin_amdgcn_s_wait_tensorcnt)
#define HAVE_TDM 1
#else
#define HAVE_TDM 0
#endif

// ---------------- helpers ----------------

__device__ __forceinline__ u16 f2bf(float f) {
  union { float f; u32 u; } x; x.f = f;
  u32 u = x.u;
  u32 r = u + 0x7FFFu + ((u >> 16) & 1u);  // round-to-nearest-even
  return (u16)(r >> 16);
}

#if HAVE_TDM
// Tensor Data Mover: 2D tile of 2-byte elements, global -> LDS (D# per ISA 8.3/8.4).
// dims are relative to the tile start; rows beyond rows_tensor are zero-filled.
__device__ __forceinline__ void tdm_load_2d(u32 lds_addr, const void* gaddr,
                                            u32 elems_tensor, u32 rows_tensor,
                                            u32 tile_w, u32 tile_h, u32 stride_elems) {
  const unsigned long long ga = (unsigned long long)gaddr;
  v4u g0;
  g0.x = 1u;                                   // count=1, user descriptor
  g0.y = lds_addr;                             // LDS byte address
  g0.z = (u32)ga;                              // global_addr[31:0]
  g0.w = (u32)(ga >> 32) | (2u << 30);         // global_addr[56:32] | type=2
  v8i g1;
  g1[0] = (int)(1u << 16);                     // workgroup_mask=0, data_size=2B
  g1[1] = (int)((elems_tensor & 0xFFFFu) << 16);                    // tensor_dim0[15:0]
  g1[2] = (int)((elems_tensor >> 16) | ((rows_tensor & 0xFFFFu) << 16));
  g1[3] = (int)((rows_tensor >> 16) | (tile_w << 16));              // tile_dim0
  g1[4] = (int)tile_h;                         // tile_dim1 (tile_dim2=0)
  g1[5] = (int)stride_elems;                   // tensor_dim0_stride[31:0]
  g1[6] = 0;
  g1[7] = 0;
  const v4i z4 = {0, 0, 0, 0};
#if __clang_major__ >= 23
  const v8i z8 = {0, 0, 0, 0, 0, 0, 0, 0};
  __builtin_amdgcn_tensor_load_to_lds(g0, g1, z4, z4, z8, 0);
#else
  __builtin_amdgcn_tensor_load_to_lds(g0, g1, z4, z4, 0);
#endif
}
__device__ __forceinline__ u32 lds_addr_of(const void* p) {
  return (u32)(unsigned long long)p;           // flat->LDS: low 32 bits (ISA 10.2)
}
#endif

// A fragment (16x32 bf16), ISA layout:
// lanes 0-15: V0..3 = K0..7, V4..7 = K16..23 ; lanes 16-31: V0..3 = K8..15, V4..7 = K24..31
__device__ __forceinline__ v16bf frag_a(const u16* lds, int stride, int m0, int k0, int lane) {
  union { u32 u[8]; v16bf v; } r;
  const u32* p = (const u32*)(lds + (m0 + (lane & 15)) * stride + k0);
  const int pb = (lane & 16) ? 4 : 0;
#pragma unroll
  for (int v = 0; v < 4; ++v) { r.u[v] = p[pb + v]; r.u[4 + v] = p[pb + 8 + v]; }
  return r.v;
}

// B fragment (32x16 bf16): lane holds column n; lanes 0-15 K0..7,K16..23, lanes 16-31 K8..15,K24..31
__device__ __forceinline__ v16bf frag_b(const u16* lds, int stride, int n0, int k0, int lane) {
  union { u32 u[8]; v16bf v; } r;
  const u32* p = (const u32*)(lds + (n0 + (lane & 15)) * stride + k0);
  const int pb = (lane & 16) ? 4 : 0;
#pragma unroll
  for (int v = 0; v < 4; ++v) { r.u[v] = p[pb + v]; r.u[4 + v] = p[pb + 8 + v]; }
  return r.v;
}

__device__ __forceinline__ v8f bf16_wmma(v16bf a, v16bf b, v8f c) {
  return __builtin_amdgcn_wmma_f32_16x16x32_bf16(false, a, false, b, (short)0, c, false, false);
}

// ---------------- kernel 1: QKV GEMM + scatter ----------------
// C(100352x1536) = xt(100352x512) * W(512x1536) + b ; scatter into q/k/v bf16 head-major
// 128x64 tile per WG, BK=64, 32x32 per wave (2x2 WMMA tiles, 8 WMMA/stage/wave)

__global__ void __launch_bounds__(256) qkv_gemm(const float* __restrict__ x,
                                                const float* __restrict__ w,
                                                const float* __restrict__ bias,
                                                u16* __restrict__ qb,
                                                u16* __restrict__ kbuf,
                                                u16* __restrict__ vb) {
  __shared__ u16 As[128 * 64];  // [m][k]
  __shared__ u16 Bs[64 * 64];   // [n][k] (transposed)
  const int tid = threadIdx.x;
  const int lane = tid & 31;
  const int wave = tid >> 5;
  const int colBase = blockIdx.x * 64;
  const int rowBase = blockIdx.y * 128;

  // A staging: each thread owns one row (128 rows, 2 threads/row), 32 consecutive k
  const int arow = tid & 127;
  const int akk = (tid >> 7) * 32;                // 0 or 32
  const int grow = rowBase + arow;
  const int gb = grow / 49;
  const int gn = grow - gb * 49;
  const float* xrow = x + (long)gb * (512L * 49) + gn;  // + k*49

  // B staging: each thread owns one column (64 cols, 4 threads/col), 16 consecutive k
  const int bcol = tid & 63;
  const int bkk = (tid >> 6) * 16;                // 0,16,32,48
  const int gcolb = colBase + bcol;

  const int m0 = (wave >> 1) * 32;
  const int n0 = (wave & 1) * 32;
  v8f acc00 = {}, acc01 = {}, acc10 = {}, acc11 = {};

  for (int k0 = 0; k0 < 512; k0 += 64) {
#pragma unroll
    for (int i = 0; i < 32; ++i)
      As[arow * 64 + akk + i] = f2bf(xrow[(long)(k0 + akk + i) * 49]);
#pragma unroll
    for (int i = 0; i < 16; ++i)
      Bs[bcol * 64 + bkk + i] = f2bf(w[(long)(k0 + bkk + i) * 1536 + gcolb]);
    if (k0 + 64 < 512) {  // prefetch next stage (global_prefetch_b8)
      __builtin_prefetch(&xrow[(long)(k0 + 64 + akk) * 49], 0, 3);
      __builtin_prefetch(&w[(long)(k0 + 64 + bkk) * 1536 + gcolb], 0, 3);
    }
    __syncthreads();
#pragma unroll
    for (int ks = 0; ks < 64; ks += 32) {
      v16bf a0 = frag_a(As, 64, m0, ks, lane);
      v16bf a1 = frag_a(As, 64, m0 + 16, ks, lane);
      v16bf b0 = frag_b(Bs, 64, n0, ks, lane);
      v16bf b1 = frag_b(Bs, 64, n0 + 16, ks, lane);
      acc00 = bf16_wmma(a0, b0, acc00);
      acc01 = bf16_wmma(a0, b1, acc01);
      acc10 = bf16_wmma(a1, b0, acc10);
      acc11 = bf16_wmma(a1, b1, acc11);
    }
    __syncthreads();
  }

  const int hi = (lane & 16) ? 8 : 0;
  const int nl = lane & 15;
#pragma unroll
  for (int mi = 0; mi < 2; ++mi) {
#pragma unroll
    for (int ni = 0; ni < 2; ++ni) {
      const v8f acc = mi ? (ni ? acc11 : acc10) : (ni ? acc01 : acc00);
      const int gcol = colBase + n0 + ni * 16 + nl;  // = d*48 + h*3 + s
      const int s  = gcol % 3;
      const int hh = (gcol / 3) & 15;
      const int dd = gcol / 48;
      u16* dst = (s == 0) ? qb : (s == 1) ? kbuf : vb;
      const float sc = (s == 0) ? 0.17677669529663689f : 1.0f;  // hd^-0.5
      const float bv = bias[gcol];
#pragma unroll
      for (int v = 0; v < 8; ++v) {
        const int gm = rowBase + m0 + mi * 16 + v + hi;
        const int b = gm / 49, nr = gm - b * 49;
        const float val = (acc[v] + bv) * sc;
        dst[(((long)b * 16 + hh) * 49 + nr) * 32 + dd] = f2bf(val);
      }
    }
  }
}

// ---------------- kernel 2: windowed attention per (batch, head) ----------------

__global__ void __launch_bounds__(256) win_attn(const u16* __restrict__ qb,
                                                const u16* __restrict__ kbuf,
                                                const u16* __restrict__ vb,
                                                const float* __restrict__ bias_table,
                                                const float* __restrict__ mask,
                                                u16* __restrict__ ob) {
  __shared__ u16 Qs[64 * 32];    // [m][d], rows >=49 zero
  __shared__ u16 Ks[64 * 32];    // [n][d]
  __shared__ u16 Vt[32 * 64];    // [d][k], cols >=49 zero
  __shared__ float Sf[64 * 64];  // scores
  __shared__ u16 Ps[64 * 64];    // probs bf16

  const int tid = threadIdx.x;
  const int lane = tid & 31;
  const int wave = tid >> 5;
  const int b = blockIdx.x >> 4;
  const int h = blockIdx.x & 15;
  const long base = ((long)b * 16 + h) * 49 * 32;

#if HAVE_TDM
  // TDM stages Q and K (rows >=49 auto zero-filled via tensor OOB); V transposed manually.
  if (wave == 0) {
    tdm_load_2d(lds_addr_of(Qs), qb + base, 32, 49, 32, 64, 32);
    tdm_load_2d(lds_addr_of(Ks), kbuf + base, 32, 49, 32, 64, 32);
  }
#pragma unroll
  for (int i = 0; i < 8; ++i) {
    const int idx = tid + i * 256;     // 0..2047
    const int dd = idx >> 6, n = idx & 63;
    Vt[idx] = (n < 49) ? vb[base + n * 32 + dd] : (u16)0;
  }
  if (wave == 0) __builtin_amdgcn_s_wait_tensorcnt(0);
  __syncthreads();
#else
#pragma unroll
  for (int i = 0; i < 8; ++i) {
    const int idx = tid + i * 256;     // 0..2047
    const int r = idx >> 5, d = idx & 31;
    u16 qv = 0, kv = 0;
    if (r < 49) { qv = qb[base + r * 32 + d]; kv = kbuf[base + r * 32 + d]; }
    Qs[idx] = qv; Ks[idx] = kv;
    const int dd = idx >> 6, n = idx & 63;
    Vt[idx] = (n < 49) ? vb[base + n * 32 + dd] : (u16)0;
  }
  __syncthreads();
#endif

  // Phase 1: S = Q * K^T (+bias, +mask)
  {
    const int m0 = (wave >> 1) * 16;
    const int n0 = (wave & 1) * 32;
    v16bf a  = frag_a(Qs, 32, m0, 0, lane);
    v16bf b0 = frag_b(Ks, 32, n0, 0, lane);
    v16bf b1 = frag_b(Ks, 32, n0 + 16, 0, lane);
    v8f s0 = {}; v8f s1 = {};
    s0 = bf16_wmma(a, b0, s0);
    s1 = bf16_wmma(a, b1, s1);
    const int nl = lane & 15;
    const int hi = (lane & 16) ? 8 : 0;
    const float* mrow = mask + (long)(b & 63) * 49 * 49;
#pragma unroll
    for (int t = 0; t < 2; ++t) {
      const v8f sv = t ? s1 : s0;
      const int n = n0 + t * 16 + nl;
#pragma unroll
      for (int v = 0; v < 8; ++v) {
        const int m = m0 + v + hi;
        float val = sv[v];
        if (m < 49 && n < 49) {
          const int i0 = (m / 7) - (n / 7) + 6;
          const int i1 = (m % 7) - (n % 7) + 6;
          val += bias_table[(i0 * 6 + i1) * 16 + h] + mrow[m * 49 + n];
        } else {
          val = -1.0e30f;
        }
        Sf[m * 64 + n] = val;
      }
    }
  }
  __syncthreads();

  // Softmax: one row per thread (rows 0..63; padded rows harmless)
  if (tid < 64) {
    const float* row = Sf + tid * 64;
    float mx = row[0];
    for (int c = 1; c < 64; ++c) mx = fmaxf(mx, row[c]);
    float sum = 0.f;
    for (int c = 0; c < 64; ++c) sum += __expf(row[c] - mx);
    const float inv = 1.0f / sum;
    u16* prow = Ps + tid * 64;
    for (int c = 0; c < 64; ++c) prow[c] = f2bf(__expf(row[c] - mx) * inv);
  }
  __syncthreads();

  // Phase 2: O = P * V (K=64 -> two WMMA steps)
  {
    const int m0 = (wave >> 1) * 16;
    const int d0 = (wave & 1) * 16;
    v8f o = {};
#pragma unroll
    for (int ks = 0; ks < 64; ks += 32) {
      v16bf a  = frag_a(Ps, 64, m0, ks, lane);
      v16bf bb = frag_b(Vt, 64, d0, ks, lane);
      o = bf16_wmma(a, bb, o);
    }
    const int nl = lane & 15;
    const int hi = (lane & 16) ? 8 : 0;
#pragma unroll
    for (int v = 0; v < 8; ++v) {
      const int m = m0 + v + hi;
      if (m < 49) ob[((long)(b * 49 + m)) * 512 + h * 32 + d0 + nl] = f2bf(o[v]);
    }
  }
}

// ---------------- kernel 3: output projection (+ NHWC->NCHW transpose) ----------------
// 128x64 tile per WG, BK=64, 32x32 per wave; A tile staged by TDM (bf16 source)

__global__ void __launch_bounds__(256) proj_gemm(const u16* __restrict__ ain,
                                                 const float* __restrict__ w,
                                                 const float* __restrict__ bias,
                                                 float* __restrict__ out) {
  __shared__ u16 As[128 * 64];
  __shared__ u16 Bs[64 * 64];
  const int tid = threadIdx.x;
  const int lane = tid & 31;
  const int wave = tid >> 5;
  const int colBase = blockIdx.x * 64;
  const int rowBase = blockIdx.y * 128;

#if !HAVE_TDM
  const int arow = tid & 127;
  const int akp = (tid >> 7) * 16;               // dword offset 0 or 16
  const u32* arow32 = (const u32*)(ain + (long)(rowBase + arow) * 512);
  u32* As32 = (u32*)As;
#endif

  const int bcol = tid & 63;
  const int bkk = (tid >> 6) * 16;
  const int gcolb = colBase + bcol;

  const int m0 = (wave >> 1) * 32;
  const int n0 = (wave & 1) * 32;
  v8f acc00 = {}, acc01 = {}, acc10 = {}, acc11 = {};

  for (int k0 = 0; k0 < 512; k0 += 64) {
#if HAVE_TDM
    if (wave == 0)
      tdm_load_2d(lds_addr_of(As), ain + (long)rowBase * 512 + k0,
                  512 - k0, 100352 - rowBase, 64, 128, 512);
#else
#pragma unroll
    for (int i = 0; i < 16; ++i)
      As32[arow * 32 + akp + i] = arow32[(k0 >> 1) + akp + i];
#endif
#pragma unroll
    for (int i = 0; i < 16; ++i)
      Bs[bcol * 64 + bkk + i] = f2bf(w[(long)(k0 + bkk + i) * 512 + gcolb]);
    if (k0 + 64 < 512)
      __builtin_prefetch(&w[(long)(k0 + 64 + bkk) * 512 + gcolb], 0, 3);
#if HAVE_TDM
    if (wave == 0) __builtin_amdgcn_s_wait_tensorcnt(0);
#endif
    __syncthreads();
#pragma unroll
    for (int ks = 0; ks < 64; ks += 32) {
      v16bf a0 = frag_a(As, 64, m0, ks, lane);
      v16bf a1 = frag_a(As, 64, m0 + 16, ks, lane);
      v16bf b0 = frag_b(Bs, 64, n0, ks, lane);
      v16bf b1 = frag_b(Bs, 64, n0 + 16, ks, lane);
      acc00 = bf16_wmma(a0, b0, acc00);
      acc01 = bf16_wmma(a0, b1, acc01);
      acc10 = bf16_wmma(a1, b0, acc10);
      acc11 = bf16_wmma(a1, b1, acc11);
    }
    __syncthreads();
  }

  const int hi = (lane & 16) ? 8 : 0;
  const int nl = lane & 15;
#pragma unroll
  for (int mi = 0; mi < 2; ++mi) {
#pragma unroll
    for (int ni = 0; ni < 2; ++ni) {
      const v8f acc = mi ? (ni ? acc11 : acc10) : (ni ? acc01 : acc00);
      const int gcol = colBase + n0 + ni * 16 + nl;
      const float bv = bias[gcol];
#pragma unroll
      for (int v = 0; v < 8; ++v) {
        const long gm = rowBase + m0 + mi * 16 + v + hi;
        const long bidx = gm / 49;
        const int nr = (int)(gm - bidx * 49);
        out[(bidx * 512 + gcol) * 49 + nr] = acc[v] + bv;  // (b, c, 7, 7)
      }
    }
  }
}

// ---------------- launch ----------------

extern "C" void kernel_launch(void* const* d_in, const int* in_sizes, int n_in,
                              void* d_out, int out_size, void* d_ws, size_t ws_size,
                              hipStream_t stream) {
  (void)in_sizes; (void)n_in; (void)out_size; (void)ws_size;
  const float* x          = (const float*)d_in[0];
  const float* mask       = (const float*)d_in[1];
  const float* bias_table = (const float*)d_in[2];
  const float* qkv_w      = (const float*)d_in[3];
  const float* qkv_b      = (const float*)d_in[4];
  const float* proj_w     = (const float*)d_in[5];
  const float* proj_b     = (const float*)d_in[6];
  float* out = (float*)d_out;

  const long QS = 2048L * 16 * 49 * 32;  // elements per q/k/v bf16 buffer
  u16* qb = (u16*)d_ws;
  u16* kb = qb + QS;
  u16* vb = kb + QS;
  u16* ob = vb + QS;                     // attention output, 100352 x 512 bf16

  qkv_gemm<<<dim3(1536 / 64, 100352 / 128), 256, 0, stream>>>(x, qkv_w, qkv_b, qb, kb, vb);
  win_attn<<<dim3(2048 * 16), 256, 0, stream>>>(qb, kb, vb, bias_table, mask, ob);
  proj_gemm<<<dim3(512 / 64, 100352 / 128), 256, 0, stream>>>(ob, proj_w, proj_b, out);
}
